// BilstmCrfModel_14370960573046
// MI455X (gfx1250) — compile-verified, matrix operand order checked by reading, not run on hardware
//
#include <hip/hip_runtime.h>
#include <hip/hip_bf16.h>
#include <math.h>
#include <stdint.h>

// Problem constants (match reference)
#define VOCABSZ 50000
#define EMBD    128
#define HIDD    256
#define NLBL    9
#define BB      128
#define TT      512
#define HH      128       // HID/2
#define G4      512       // 4*H

typedef __attribute__((ext_vector_type(16))) __bf16 v16bf;
typedef __attribute__((ext_vector_type(8)))  __bf16 v8bf;
typedef __attribute__((ext_vector_type(8)))  float  v8f;

__device__ __forceinline__ float sigmf(float x) {
    return 1.0f / (1.0f + __expf(-x));
}

// Async direct-to-LDS 16B copy (CDNA5 GLOBAL_LOAD_ASYNC_TO_LDS_B128, ASYNCcnt).
// lds_dst: pointer into LDS (low 32 bits of the generic address are the wave's
// LDS byte address); gsrc: per-lane global address.
__device__ __forceinline__ void async_lds_load16(void* lds_dst, const void* gsrc) {
    unsigned ldsa = (unsigned)(uintptr_t)lds_dst;
    asm volatile("global_load_async_to_lds_b128 %0, %1, off"
                 :: "v"(ldsa), "v"(gsrc) : "memory");
}
__device__ __forceinline__ void wait_asynccnt0() {
    asm volatile("s_wait_asynccnt 0x0" ::: "memory");
}

// Load a 16x32 bf16 WMMA fragment (A or B side) from a row-major [rows][128]
// bf16 array. Per CDNA5 ISA 16-bit A-matrix layout: lanes 0-15 carry K 0-7 and
// 16-23, lanes 16-31 carry K 8-15 and 24-31; row = row_base + (lane&15).
__device__ __forceinline__ v16bf frag_ld(const __bf16* mat, int row_base, int lane, int k0) {
    int r  = row_base + (lane & 15);
    int kb = k0 + ((lane >> 4) << 3);
    const __bf16* p = mat + r * 128 + kb;
    v8bf lo = *(const v8bf*)(p);
    v8bf hi = *(const v8bf*)(p + 16);
    v16bf out;
#pragma unroll
    for (int i = 0; i < 8; ++i) { out[i] = lo[i]; out[i + 8] = hi[i]; }
    return out;
}

// ---------------------------------------------------------------------------
// Kernel 1: embedding gather, fp32 table -> bf16 x buffer [B][T][EMB]
// ---------------------------------------------------------------------------
__global__ void embed_kernel(const int* __restrict__ chars,
                             const float* __restrict__ emb,
                             __bf16* __restrict__ xg) {
    size_t idx = (size_t)blockIdx.x * blockDim.x + threadIdx.x;
    if (idx >= (size_t)BB * TT * EMBD) return;
    size_t bt  = idx >> 7;          // / EMB
    int    col = (int)(idx & 127);
    int    ch  = chars[bt];
    xg[idx] = (__bf16)emb[(size_t)ch * EMBD + col];
}

// ---------------------------------------------------------------------------
// Kernel 2: LSTM recurrence. 16 blocks = 8 batch slices x 2 directions.
// Weights live in LDS (bf16) for the whole kernel; cell state c lives in VGPRs.
// x tiles are double-buffered via async direct-to-LDS copies (ASYNCcnt) so the
// global fetch of x_{t+1} overlaps the 32 WMMAs of step t.
// Dynamic LDS: 2*512*128*2 + 512*4 + 3*16*128*2 = 276480 bytes (< 320KB WGP LDS)
// ---------------------------------------------------------------------------
__global__ void __launch_bounds__(256)
lstm_kernel(const __bf16* __restrict__ xg,
            const float* __restrict__ WihF, const float* __restrict__ WhhF,
            const float* __restrict__ bihF, const float* __restrict__ bhhF,
            const float* __restrict__ WihB, const float* __restrict__ WhhB,
            const float* __restrict__ bihB, const float* __restrict__ bhhB,
            __bf16* __restrict__ hF, __bf16* __restrict__ hB) {
    extern __shared__ char smem[];
    __bf16* wih  = (__bf16*)smem;             // [512][128]
    __bf16* whh  = wih + G4 * HH;             // [512][128]
    float*  bias = (float*)(whh + G4 * HH);   // [512]
    __bf16* xs0  = (__bf16*)(bias + G4);      // [16][128] x ping
    __bf16* xs1  = xs0 + 16 * EMBD;           // [16][128] x pong
    __bf16* hs   = xs1 + 16 * EMBD;           // [16][128] current h_t slice

    const int tid   = threadIdx.x;
    const int lane  = tid & 31;
    const int wv    = tid >> 5;               // wave 0..7 -> hidden cols [wv*16, wv*16+16)
    const int dir   = blockIdx.x >> 3;        // 0 = forward, 1 = backward
    const int slice = blockIdx.x & 7;         // batch rows [slice*16, slice*16+16)

    const float* Wih = dir ? WihB : WihF;
    const float* Whh = dir ? WhhB : WhhF;
    const float* bih = dir ? bihB : bihF;
    const float* bhh = dir ? bhhB : bhhF;
    __bf16* hout = dir ? hB : hF;

    // Stage weights fp32->bf16 into LDS once; precombine biases.
    for (int i = tid; i < G4 * HH; i += 256) {
        wih[i] = (__bf16)Wih[i];
        whh[i] = (__bf16)Whh[i];
    }
    for (int i = tid; i < G4; i += 256) bias[i] = bih[i] + bhh[i];
    for (int i = tid; i < 16 * HH; i += 256) hs[i] = (__bf16)0.0f;

    // Per-thread 16B slice of the x tile: 256 threads x 16B = 16x128 bf16.
    const int xrow = (tid * 8) >> 7;          // 0..15
    const int xcol = (tid * 8) & 127;

    // Preload x for t=0 into the ping buffer.
    {
        const int t0 = dir ? (TT - 1) : 0;
        const __bf16* src =
            xg + ((size_t)(slice * 16 + xrow) * TT + t0) * EMBD + xcol;
        async_lds_load16(xs0 + xrow * EMBD + xcol, src);
    }
    wait_asynccnt0();
    __syncthreads();   // weights, bias, hs=0, xs0 all visible

    const int j = wv * 16 + (lane & 15);      // hidden column owned by this lane
    v8f c = {};                               // cell state: 8 rows of column j

#pragma unroll 1
    for (int t = 0; t < TT; ++t) {
        const int tc = dir ? (TT - 1 - t) : t;
        __bf16* xcur = (t & 1) ? xs1 : xs0;

        // Kick off the async copy of x_{t+1} into the other buffer; it lands
        // while this step's WMMAs run. Target buffer was last read at t-1.
        if (t + 1 < TT) {
            const int tn = dir ? (TT - 2 - t) : (t + 1);
            __bf16* xnxt = (t & 1) ? xs0 : xs1;
            const __bf16* src =
                xg + ((size_t)(slice * 16 + xrow) * TT + tn) * EMBD + xcol;
            async_lds_load16(xnxt + xrow * EMBD + xcol, src);
        }

        // Gate accumulators: i, f, g, o -- init with combined bias.
        v8f acc[4];
#pragma unroll
        for (int g = 0; g < 4; ++g) {
            float bv = bias[g * HH + j];
            v8f a;
#pragma unroll
            for (int q = 0; q < 8; ++q) a[q] = bv;
            acc[g] = a;
        }

        // gates(16x512) = x_t(16x128) @ Wih^T + h(16x128) @ Whh^T
#pragma unroll
        for (int kc = 0; kc < 4; ++kc) {
            const int k0 = kc * 32;
            v16bf ax = frag_ld(xcur, 0, lane, k0);
            v16bf ah = frag_ld(hs, 0, lane, k0);
#pragma unroll
            for (int g = 0; g < 4; ++g) {
                const int nb = g * HH + wv * 16;
                v16bf bx = frag_ld(wih, nb, lane, k0);
                acc[g] = __builtin_amdgcn_wmma_f32_16x16x32_bf16(
                    false, ax, false, bx, (short)0, acc[g], false, false);
                v16bf bh = frag_ld(whh, nb, lane, k0);
                acc[g] = __builtin_amdgcn_wmma_f32_16x16x32_bf16(
                    false, ah, false, bh, (short)0, acc[g], false, false);
            }
        }
        __syncthreads();   // all LDS reads of xcur/hs done before hs rewrite

        // Elementwise LSTM cell update; C-layout: VGPR r, lanes0-15 -> M=r,
        // lanes16-31 -> M=8+r, N = lane&15.
#pragma unroll
        for (int r = 0; r < 8; ++r) {
            float iv = sigmf(acc[0][r]);
            float fv = sigmf(acc[1][r]);
            float gv = tanhf(acc[2][r]);
            float ov = sigmf(acc[3][r]);
            float cn = fv * c[r] + iv * gv;
            c[r] = cn;
            float hv = ov * tanhf(cn);
            int   m  = ((lane >> 4) << 3) + r;
            __bf16 hb16 = (__bf16)hv;
            hs[m * HH + j] = hb16;
            hout[((size_t)(slice * 16 + m) * TT + tc) * HH + j] = hb16;
        }

        wait_asynccnt0();  // this wave's x_{t+1} copy has landed in LDS
        __syncthreads();   // new hs + next x buffer visible to all waves
    }
}

// ---------------------------------------------------------------------------
// Kernel 3: emissions = [hf, hb] @ W_out^T + b_out  -> (B*T, 9) fp32
// N=9 is far below a WMMA tile; VALU dot products are the right tool.
// ---------------------------------------------------------------------------
__global__ void emis_kernel(const __bf16* __restrict__ hF,
                            const __bf16* __restrict__ hB,
                            const float* __restrict__ Wout,
                            const float* __restrict__ bout,
                            float* __restrict__ em) {
    int idx = blockIdx.x * blockDim.x + threadIdx.x;
    if (idx >= BB * TT * NLBL) return;
    int bt = idx / NLBL;
    int l  = idx - bt * NLBL;
    const __bf16* pf = hF + (size_t)bt * HH;
    const __bf16* pb = hB + (size_t)bt * HH;
    const float*  w  = Wout + l * HIDD;
    float s = bout[l];
#pragma unroll 4
    for (int k = 0; k < HH; ++k) s += (float)pf[k] * w[k];
#pragma unroll 4
    for (int k = 0; k < HH; ++k) s += (float)pb[k] * w[HH + k];
    em[idx] = s;
}

// ---------------------------------------------------------------------------
// Kernel 4: CRF sum-NLL. One thread per batch element (128 threads, 1 block),
// sequential forward algorithm over T with a 9x9 logsumexp; LDS reduction.
// ---------------------------------------------------------------------------
__global__ void __launch_bounds__(128)
crf_kernel(const int* __restrict__ chars, const int* __restrict__ labels,
           const float* __restrict__ em, const float* __restrict__ startT,
           const float* __restrict__ endT, const float* __restrict__ trans,
           float* __restrict__ out) {
    __shared__ float s_tr[NLBL * NLBL];
    __shared__ float s_st[NLBL];
    __shared__ float s_en[NLBL];
    __shared__ float s_red[128];

    int tid = threadIdx.x;
    if (tid < NLBL * NLBL) s_tr[tid] = trans[tid];
    if (tid < NLBL) { s_st[tid] = startT[tid]; s_en[tid] = endT[tid]; }
    __syncthreads();

    const int b = tid;
    const int*   ch = chars  + b * TT;
    const int*   lb = labels + b * TT;
    const float* eb = em + (size_t)b * TT * NLBL;

    float alpha[NLBL];
    int   lab0 = lb[0];
    float m0   = (ch[0] != 0) ? 1.0f : 0.0f;
    float score = s_st[lab0] + eb[lab0] * m0;
#pragma unroll
    for (int l = 0; l < NLBL; ++l) alpha[l] = s_st[l] + eb[l];
    int cnt = (ch[0] != 0) ? 1 : 0;

#pragma unroll 1
    for (int t = 1; t < TT; ++t) {
        const float* e = eb + t * NLBL;
        bool m = (ch[t] != 0);
        float nxt[NLBL];
#pragma unroll
        for (int jj = 0; jj < NLBL; ++jj) {
            float mx = -1e30f;
#pragma unroll
            for (int i = 0; i < NLBL; ++i)
                mx = fmaxf(mx, alpha[i] + s_tr[i * NLBL + jj]);
            float s = 0.0f;
#pragma unroll
            for (int i = 0; i < NLBL; ++i)
                s += __expf(alpha[i] + s_tr[i * NLBL + jj] - mx);
            nxt[jj] = mx + __logf(s) + e[jj];
        }
        if (m) {
#pragma unroll
            for (int jj = 0; jj < NLBL; ++jj) alpha[jj] = nxt[jj];
            int lab = lb[t];
            score += e[lab] + s_tr[lb[t - 1] * NLBL + lab];
            cnt++;
        }
    }
    int lastTag = lb[cnt > 0 ? cnt - 1 : 0];
    score += s_en[lastTag];

    float mx = -1e30f;
#pragma unroll
    for (int l = 0; l < NLBL; ++l) mx = fmaxf(mx, alpha[l] + s_en[l]);
    float s = 0.0f;
#pragma unroll
    for (int l = 0; l < NLBL; ++l) s += __expf(alpha[l] + s_en[l] - mx);
    float logZ = mx + __logf(s);

    s_red[tid] = -(score - logZ);
    __syncthreads();
    for (int off = 64; off > 0; off >>= 1) {
        if (tid < off) s_red[tid] += s_red[tid + off];
        __syncthreads();
    }
    if (tid == 0) out[0] = s_red[0];
}

// ---------------------------------------------------------------------------
// Launch
// ---------------------------------------------------------------------------
extern "C" void kernel_launch(void* const* d_in, const int* in_sizes, int n_in,
                              void* d_out, int out_size, void* d_ws, size_t ws_size,
                              hipStream_t stream) {
    const int*   chars  = (const int*)d_in[0];
    const int*   labels = (const int*)d_in[1];
    const float* emb    = (const float*)d_in[2];
    const float* WihF   = (const float*)d_in[3];
    const float* WhhF   = (const float*)d_in[4];
    const float* bihF   = (const float*)d_in[5];
    const float* bhhF   = (const float*)d_in[6];
    const float* WihB   = (const float*)d_in[7];
    const float* WhhB   = (const float*)d_in[8];
    const float* bihB   = (const float*)d_in[9];
    const float* bhhB   = (const float*)d_in[10];
    const float* Wout   = (const float*)d_in[11];
    const float* bout   = (const float*)d_in[12];
    const float* startT = (const float*)d_in[13];
    const float* endT   = (const float*)d_in[14];
    const float* trans  = (const float*)d_in[15];

    // Workspace layout (bytes):
    //   xg : B*T*EMB bf16 = 16,777,216
    //   hF : B*T*H  bf16  = 16,777,216
    //   hB : B*T*H  bf16  = 16,777,216
    //   em : B*T*NL fp32  =  2,359,296   (total ~52.7 MB)
    char* ws = (char*)d_ws;
    __bf16* xg = (__bf16*)(ws);
    __bf16* hF = (__bf16*)(ws + 16777216);
    __bf16* hB = (__bf16*)(ws + 33554432);
    float*  em = (float*)(ws + 50331648);
    float*  out = (float*)d_out;

    {
        size_t n = (size_t)BB * TT * EMBD;
        embed_kernel<<<(unsigned)((n + 255) / 256), 256, 0, stream>>>(chars, emb, xg);
    }
    {
        // LDS: wih + whh (bf16) + bias(f32) + xs ping/pong + hs
        size_t shmem = (size_t)2 * G4 * HH * 2 + (size_t)G4 * 4
                     + (size_t)3 * 16 * EMBD * 2;  // 276480
        lstm_kernel<<<16, 256, (unsigned)shmem, stream>>>(
            xg, WihF, WhhF, bihF, bhhF, WihB, WhhB, bihB, bhhB, hF, hB);
    }
    {
        int n = BB * TT * NLBL;
        emis_kernel<<<(n + 255) / 256, 256, 0, stream>>>(hF, hB, Wout, bout, em);
    }
    crf_kernel<<<1, 128, 0, stream>>>(chars, labels, em, startT, endT, trans, out);
}
